// MultiHeadAttention_16784732192805
// MI455X (gfx1250) — matrix operand order repaired and checked by reading, not compiled, not run on hardware
//
#include <hip/hip_runtime.h>
#include <hip/hip_bf16.h>

typedef __attribute__((ext_vector_type(16))) __bf16 v16bf;
typedef __attribute__((ext_vector_type(8)))  float  v8f;
typedef __attribute__((ext_vector_type(4)))  int    v4i;

#define DMODEL 1024
#define NHEAD  16
#define HDIM   64
#define NB     4
#define SEQ    2048

#define KPAD 72   // lds k row stride (bf16): 144B, 16B-aligned, conflict-free
#define VPAD 40   // lds v row stride (bf16): 80B, 16B-aligned, conflict-free

#define AS3 __attribute__((address_space(3)))
#define AS1 __attribute__((address_space(1)))

#if defined(__AMDGCN__) && __has_builtin(__builtin_amdgcn_global_load_async_to_lds_b128)
#define HAS_ASYNC_LDS 1
#else
#define HAS_ASYNC_LDS 0
#endif

#if __has_builtin(__builtin_amdgcn_s_wait_asynccnt)
#define WAIT_ASYNC(n) __builtin_amdgcn_s_wait_asynccnt(n)
#elif HAS_ASYNC_LDS
#define WAIT_ASYNC(n) asm volatile("s_wait_asynccnt %0" :: "i"(n) : "memory")
#else
#define WAIT_ASYNC(n) do {} while (0)
#endif

#define WMMA_BF16(A,B,C) \
  __builtin_amdgcn_wmma_f32_16x16x32_bf16(false,(A),false,(B),(short)0,(C),false,false)

union FragU { uint4 u[2]; v16bf v; };
union PkU   { __bf16 h[2]; unsigned u; };

__device__ __forceinline__ v16bf frag_pair(const __bf16* p0, const __bf16* p1) {
  FragU f;
  f.u[0] = *(const uint4*)p0;
  f.u[1] = *(const uint4*)p1;
  return f.v;
}
__device__ __forceinline__ v16bf frag16(const __bf16* p) {
  FragU f;
  f.u[0] = *(const uint4*)p;
  f.u[1] = *(const uint4*)(p + 8);
  return f.v;
}
__device__ __forceinline__ unsigned pk2(float a, float b) {
  PkU p; p.h[0] = (__bf16)a; p.h[1] = (__bf16)b; return p.u;
}

// ---------------------------------------------------------------------------
// QKV projection: out = X[8192,1024] @ W[1024,1024] + bias, stored bf16.
// MODE 0: q, scaled 1/sqrt(64), layout [B][H][L][64]
// MODE 1: k,                     layout [B][H][L][64]
// MODE 2: v, stored transposed   layout [B][H][64][L]
// 256 threads (8 waves). Tile 128(M) x 64(N), K-steps of 32.
// ---------------------------------------------------------------------------
template <int MODE>
__global__ __launch_bounds__(256, 1)
void qkv_proj_kernel(const float* __restrict__ X, const float* __restrict__ W,
                     const float* __restrict__ bias, __bf16* __restrict__ out)
{
  __shared__ __align__(16) __bf16 lds_a[128][40];  // [m][k], 80B stride
  __shared__ __align__(16) __bf16 lds_b[64][40];   // [n][k] (transposed W tile)

  const int tid  = threadIdx.x;
  const int wv   = tid >> 5;
  const int lane = tid & 31;
  const int hf   = lane >> 4;
  const int ln   = lane & 15;
  const int m0   = blockIdx.x * 128;
  const int n0   = blockIdx.y * 64;

  v8f acc0 = {}, acc1 = {}, acc2 = {}, acc3 = {};

  for (int ks = 0; ks < DMODEL; ks += 32) {
    // A tile: 128x32 fp32 -> bf16 LDS (16 floats / thread, packed b32 stores)
#pragma unroll
    for (int j = 0; j < 4; ++j) {
      int idx = tid * 16 + j * 4;
      int row = idx >> 5, col = idx & 31;
      float4 v = *(const float4*)&X[(size_t)(m0 + row) * DMODEL + ks + col];
      *(unsigned*)&lds_a[row][col]     = pk2(v.x, v.y);
      *(unsigned*)&lds_a[row][col + 2] = pk2(v.z, v.w);
    }
    // B tile: 32x64 fp32 -> bf16 LDS transposed [n][k] (8 floats / thread)
#pragma unroll
    for (int j = 0; j < 2; ++j) {
      int idx = tid * 8 + j * 4;
      int kr = idx >> 6, nc = idx & 63;
      float4 v = *(const float4*)&W[(size_t)(ks + kr) * DMODEL + n0 + nc];
      lds_b[nc + 0][kr] = (__bf16)v.x;
      lds_b[nc + 1][kr] = (__bf16)v.y;
      lds_b[nc + 2][kr] = (__bf16)v.z;
      lds_b[nc + 3][kr] = (__bf16)v.w;
    }
    __syncthreads();

    v16bf af = frag16(&lds_a[wv * 16 + ln][hf * 16]);
    acc0 = WMMA_BF16(af, frag16(&lds_b[ 0 + ln][hf * 16]), acc0);
    acc1 = WMMA_BF16(af, frag16(&lds_b[16 + ln][hf * 16]), acc1);
    acc2 = WMMA_BF16(af, frag16(&lds_b[32 + ln][hf * 16]), acc2);
    acc3 = WMMA_BF16(af, frag16(&lds_b[48 + ln][hf * 16]), acc3);
    __syncthreads();
  }

  v8f accs[4] = {acc0, acc1, acc2, acc3};
#pragma unroll
  for (int nt = 0; nt < 4; ++nt) {
#pragma unroll
    for (int r = 0; r < 8; ++r) {
      int row = wv * 16 + hf * 8 + r;
      int gm  = m0 + row;
      int col = n0 + nt * 16 + ln;
      float val = accs[nt][r] + bias[col];
      int bb = gm >> 11, l = gm & (SEQ - 1);
      int hh = col >> 6, d = col & (HDIM - 1);
      size_t off;
      if (MODE == 2)
        off = (((size_t)(bb * NHEAD + hh)) * HDIM + d) * SEQ + l;   // v^T
      else
        off = (((size_t)(bb * NHEAD + hh)) * SEQ + l) * HDIM + d;   // q / k
      if (MODE == 0) val *= 0.125f;  // 1/sqrt(dk) folded into q
      out[off] = (__bf16)val;
    }
  }
}

// ---------------------------------------------------------------------------
// Stage one 32-key tile of K [32][64] and V^T [64][32] (bf16) into LDS.
// 256 threads x one 16B chunk per buffer -> async global->LDS if available.
// ---------------------------------------------------------------------------
__device__ __forceinline__ void stage_kv(int buf, int kk, int tid,
    const __bf16* __restrict__ kp, const __bf16* __restrict__ vp,
    __bf16 lk[2][32][KPAD], __bf16 lv[2][64][VPAD])
{
  const int kr = tid >> 3, kc = (tid & 7) * 8;      // k tile: 32 rows x 8 chunks
  const __bf16* gk = kp + (size_t)(kk + kr) * HDIM + kc;
  __bf16* dk = &lk[buf][kr][kc];
  const int vr = tid >> 2, vc = (tid & 3) * 8;      // v tile: 64 rows x 4 chunks
  const __bf16* gv = vp + (size_t)vr * SEQ + kk + vc;
  __bf16* dv = &lv[buf][vr][vc];
#if HAS_ASYNC_LDS
  __builtin_amdgcn_global_load_async_to_lds_b128((AS1 v4i*)gk, (AS3 v4i*)dk, 0, 0);
  __builtin_amdgcn_global_load_async_to_lds_b128((AS1 v4i*)gv, (AS3 v4i*)dv, 0, 0);
#else
  *(uint4*)dk = *(const uint4*)gk;
  *(uint4*)dv = *(const uint4*)gv;
#endif
}

// ---------------------------------------------------------------------------
// Attention: per (b,h), 8 waves x 16 q-rows = 128 q rows per block.
// Double-buffered async LDS staging of K/V^T tiles shared by all 8 waves.
// Raw exp => single pass, normalize at the end.
// ---------------------------------------------------------------------------
__global__ __launch_bounds__(256, 1)
void attn_kernel(const __bf16* __restrict__ qb, const __bf16* __restrict__ kb,
                 const __bf16* __restrict__ vb, const int* __restrict__ mask,
                 float* __restrict__ out)
{
  __shared__ __align__(16) __bf16 lds_k[2][32][KPAD];
  __shared__ __align__(16) __bf16 lds_v[2][64][VPAD];
  __shared__ __align__(16) __bf16 lds_p[8][16][40];   // per-wave P staging

  const int tid  = threadIdx.x;
  const int wv   = tid >> 5;
  const int lane = tid & 31;
  const int hf   = lane >> 4;
  const int ln   = lane & 15;
  const int bh   = blockIdx.x;
  const int b    = bh >> 4;
  const int h    = bh & (NHEAD - 1);
  const int q0   = blockIdx.y * 128 + wv * 16;

  const __bf16* qp = qb + (size_t)bh * SEQ * HDIM;
  const __bf16* kp = kb + (size_t)bh * SEQ * HDIM;
  const __bf16* vp = vb + (size_t)bh * HDIM * SEQ;
  const int*    mp = mask + b * SEQ;

  const __bf16* qrow = qp + (size_t)(q0 + ln) * HDIM;
  v16bf aq0 = frag16(qrow + hf * 16);       // dk 0..31
  v16bf aq1 = frag16(qrow + 32 + hf * 16);  // dk 32..63

  v8f o0 = {}, o1 = {}, o2 = {}, o3 = {};
  float rs[8] = {0.f, 0.f, 0.f, 0.f, 0.f, 0.f, 0.f, 0.f};

  stage_kv(0, 0, tid, kp, vp, lds_k, lds_v);

  for (int kk = 0; kk < SEQ; kk += 32) {
    const int cur = (kk >> 5) & 1;
    const bool has_next = (kk + 32) < SEQ;

    __syncthreads();                     // prior readers of buf cur^1 are done
    if (has_next) stage_kv(cur ^ 1, kk + 32, tid, kp, vp, lds_k, lds_v);
    if (has_next) { WAIT_ASYNC(2); } else { WAIT_ASYNC(0); }
    __syncthreads();                     // buf cur fully arrived for all waves

    float mk0 = (float)mp[kk + ln];
    float mk1 = (float)mp[kk + 16 + ln];

    v8f s0 = {};
    s0 = WMMA_BF16(aq0, frag16(&lds_k[cur][ln][hf * 16]),      s0);
    s0 = WMMA_BF16(aq1, frag16(&lds_k[cur][ln][32 + hf * 16]), s0);
    v8f s1 = {};
    s1 = WMMA_BF16(aq0, frag16(&lds_k[cur][16 + ln][hf * 16]),      s1);
    s1 = WMMA_BF16(aq1, frag16(&lds_k[cur][16 + ln][32 + hf * 16]), s1);

#pragma unroll
    for (int r = 0; r < 8; ++r) {
      float p0 = __expf(s0[r]) * mk0;
      float p1 = __expf(s1[r]) * mk1;
      rs[r] += p0 + p1;
      lds_p[wv][hf * 8 + r][ln]      = (__bf16)p0;   // keys kk..kk+15
      lds_p[wv][hf * 8 + r][16 + ln] = (__bf16)p1;   // keys kk+16..kk+31
    }
    // same-wave LDS RAW: DS pipe is in-order; fence stops compiler reordering
    asm volatile("s_wait_dscnt 0" ::: "memory");

    v16bf ap = frag16(&lds_p[wv][ln][hf * 16]);
    o0 = WMMA_BF16(ap, frag16(&lds_v[cur][ 0 + ln][hf * 16]), o0);
    o1 = WMMA_BF16(ap, frag16(&lds_v[cur][16 + ln][hf * 16]), o1);
    o2 = WMMA_BF16(ap, frag16(&lds_v[cur][32 + ln][hf * 16]), o2);
    o3 = WMMA_BF16(ap, frag16(&lds_v[cur][48 + ln][hf * 16]), o3);
  }

  // row-sum butterfly inside each 16-lane group (xor masks < 16 keep hf fixed)
  float inv[8];
#pragma unroll
  for (int r = 0; r < 8; ++r) {
    float s = rs[r];
    s += __shfl_xor(s, 1);
    s += __shfl_xor(s, 2);
    s += __shfl_xor(s, 4);
    s += __shfl_xor(s, 8);
    inv[r] = 1.0f / (s + 1e-8f);
  }

  v8f oo[4] = {o0, o1, o2, o3};
#pragma unroll
  for (int nt = 0; nt < 4; ++nt) {
#pragma unroll
    for (int r = 0; r < 8; ++r) {
      int row = hf * 8 + r;
      int l   = q0 + row;
      out[((size_t)(b * SEQ + l)) * DMODEL + h * HDIM + nt * 16 + ln] =
          oo[nt][r] * inv[r];
    }
  }
}

// ---------------------------------------------------------------------------
extern "C" void kernel_launch(void* const* d_in, const int* in_sizes, int n_in,
                              void* d_out, int out_size, void* d_ws, size_t ws_size,
                              hipStream_t stream) {
  (void)in_sizes; (void)n_in; (void)out_size; (void)ws_size;
  const float* Q    = (const float*)d_in[0];
  const float* K    = (const float*)d_in[1];
  const float* V    = (const float*)d_in[2];
  const int*   msk  = (const int*)d_in[3];
  const float* Wq   = (const float*)d_in[4];
  const float* bq   = (const float*)d_in[5];
  const float* Wk   = (const float*)d_in[6];
  const float* bk   = (const float*)d_in[7];
  const float* Wv   = (const float*)d_in[8];
  const float* bv   = (const float*)d_in[9];
  float* out = (float*)d_out;

  const size_t per = (size_t)NB * NHEAD * SEQ * HDIM;  // 8 Mi elems (16 MB bf16)
  __bf16* qws = (__bf16*)d_ws;
  __bf16* kws = qws + per;
  __bf16* vws = kws + per;

  dim3 pgrid(64, 16);   // 8192/128 x 1024/64
  qkv_proj_kernel<0><<<pgrid, 256, 0, stream>>>(Q, Wq, bq, qws);
  qkv_proj_kernel<1><<<pgrid, 256, 0, stream>>>(K, Wk, bk, kws);
  qkv_proj_kernel<2><<<pgrid, 256, 0, stream>>>(V, Wv, bv, vws);

  dim3 agrid(NB * NHEAD, SEQ / 128);  // 64 x 16
  attn_kernel<<<agrid, 256, 0, stream>>>(qws, kws, vws, msk, out);
}
